// IMUEncoder_80315888435269
// MI455X (gfx1250) — compile-verified, matrix operand order checked by reading, not run on hardware
//
#include <hip/hip_runtime.h>
#include <hip/hip_bf16.h>
#include <math.h>

// Problem dims (match reference)
#define B_   8
#define T_   1024
#define CIN_ 6
#define D_   128
#define L_   2
#define DI_  256
#define DS_  32
#define DC_  4
#define DTR_ 8
#define XPW_ 72          // DTR + 2*DS actual width of xproj output
#define XP_  128         // padded xproj width (multiple of 64 for WMMA tiles)
#define BT_  (B_*T_)

typedef __attribute__((ext_vector_type(16))) __bf16 v16bf;
typedef __attribute__((ext_vector_type(8)))  __bf16 v8bf;
typedef __attribute__((ext_vector_type(8)))  float  v8f;

__device__ __forceinline__ float sigmoidf_(float x) { return 1.0f / (1.0f + __expf(-x)); }
__device__ __forceinline__ float siluf_(float x)    { return x * sigmoidf_(x); }
__device__ __forceinline__ float softplusf_(float x){ return (x > 20.0f) ? x : log1pf(__expf(x)); }

// ---------------------------------------------------------------------------
// Frontend: Conv1d(CIN->D, k=3, pad=1) + BN(eval) + ReLU.  One thread per (b,t,d).
// ---------------------------------------------------------------------------
__global__ __launch_bounds__(256) void frontend_kernel(
    const float* __restrict__ x,   // (B,T,CIN)
    const float* __restrict__ fw,  // (D,CIN,3)
    const float* __restrict__ fb,  // (D)
    const float* __restrict__ bng, // (D)
    const float* __restrict__ bnb, // (D)
    float* __restrict__ h)         // (B,T,D)
{
  int id = blockIdx.x * 256 + threadIdx.x;   // over B*T*D
  int d  = id & (D_ - 1);
  int bt = id / D_;
  int t  = bt & (T_ - 1);
  int b  = bt / T_;
  float acc = fb[d];
#pragma unroll
  for (int j = 0; j < 3; ++j) {
    int tt = t - 1 + j;
    if (tt < 0 || tt >= T_) continue;
    const float* xp = x + ((size_t)b * T_ + tt) * CIN_;
    const float* wp = fw + (size_t)d * CIN_ * 3 + j;
#pragma unroll
    for (int c = 0; c < CIN_; ++c) acc += xp[c] * wp[c * 3];
  }
  float scale = bng[d] * rsqrtf(1.0f + 1e-5f);
  acc = acc * scale + bnb[d];
  h[id] = fmaxf(acc, 0.0f);
}

// ---------------------------------------------------------------------------
// RMSNorm a row of D=128 (one wave per row, 4 elems/lane), write bf16.
// ---------------------------------------------------------------------------
__global__ __launch_bounds__(256) void rmsnorm_bf16_kernel(
    const float* __restrict__ x, const float* __restrict__ w,
    __bf16* __restrict__ out)
{
  int row  = (blockIdx.x * 256 + threadIdx.x) >> 5;
  int lane = threadIdx.x & 31;
  const float4 v = ((const float4*)(x + (size_t)row * D_))[lane];
  float ss = v.x * v.x + v.y * v.y + v.z * v.z + v.w * v.w;
#pragma unroll
  for (int off = 16; off; off >>= 1) ss += __shfl_xor(ss, off, 32);
  float s = rsqrtf(ss / (float)D_ + 1e-6f);
  int c = lane * 4;
  __bf16* op = out + (size_t)row * D_ + c;
  op[0] = (__bf16)(v.x * w[c + 0] * s);
  op[1] = (__bf16)(v.y * w[c + 1] * s);
  op[2] = (__bf16)(v.z * w[c + 2] * s);
  op[3] = (__bf16)(v.w * w[c + 3] * s);
}

// Final RMSNorm -> f32 output
__global__ __launch_bounds__(256) void rmsnorm_out_kernel(
    const float* __restrict__ x, const float* __restrict__ w,
    float* __restrict__ out)
{
  int row  = (blockIdx.x * 256 + threadIdx.x) >> 5;
  int lane = threadIdx.x & 31;
  const float4 v = ((const float4*)(x + (size_t)row * D_))[lane];
  float ss = v.x * v.x + v.y * v.y + v.z * v.z + v.w * v.w;
#pragma unroll
  for (int off = 16; off; off >>= 1) ss += __shfl_xor(ss, off, 32);
  float s = rsqrtf(ss / (float)D_ + 1e-6f);
  int c = lane * 4;
  float4 o;
  o.x = v.x * w[c + 0] * s;
  o.y = v.y * w[c + 1] * s;
  o.z = v.z * w[c + 2] * s;
  o.w = v.w * w[c + 3] * s;
  ((float4*)(out + (size_t)row * D_))[lane] = o;
}

// ---------------------------------------------------------------------------
// Weight convert+transpose: W (K x N row-major f32) -> WT (Npad x K bf16),
// zero padding rows n >= N.  Grid covers Npad*K.
// ---------------------------------------------------------------------------
__global__ __launch_bounds__(256) void transpose_bf16_kernel(
    const float* __restrict__ w, __bf16* __restrict__ wt, int K, int N)
{
  int id = blockIdx.x * 256 + threadIdx.x;
  int n = id / K, k = id - n * K;
  float v = (n < N) ? w[(size_t)k * N + n] : 0.0f;
  wt[id] = (__bf16)v;
}

// ---------------------------------------------------------------------------
// WMMA bf16 GEMM: C(MxN) = A(MxK) * BT(NxK)^T [+ C if accumulate].
// Block = 256 threads = 8 waves; wave tile 16(M) x 64(N); grid (M/128, N/64).
// All 8 waves share one 64xK B tile -> staged once into LDS via gfx1250
// async LDS-DMA (GLOBAL_LOAD_ASYNC_TO_LDS_B128, ASYNCcnt), then read back as
// ds_load_b128 fragments.  A fragments stream directly from global.
// Fragment layout per CDNA5 ISA 16-bit A 16x32: lanes 0-15 hold K{0..7,16..23},
// lanes 16-31 hold K{8..15,24..31} -> two contiguous 16B loads per fragment.
// ---------------------------------------------------------------------------
__global__ __launch_bounds__(256) void wmma_gemm_kernel(
    const __bf16* __restrict__ A,
    const __bf16* __restrict__ BT,
    float* __restrict__ C,
    int K, int N, int accumulate)
{
  __shared__ __bf16 sB[64 * 256];              // supports K up to 256 (32 KB)

  int tid = threadIdx.x;
  int n0  = blockIdx.y * 64;

  // ---- stage B tile (64 rows x K) into LDS, 16 bytes per async op ----
  {
    int kc = K >> 3;                           // 16B chunks per row
    int chunks = kc << 6;                      // 64 rows
    for (int c = tid; c < chunks; c += 256) {
      int row = c / kc;
      int col = (c - row * kc) << 3;           // element offset within row
      const __bf16* src = BT + (size_t)(n0 + row) * K + col;
      unsigned dst = (unsigned)(size_t)(const void*)(sB + row * K + col);
      asm volatile("global_load_async_to_lds_b128 %0, %1, off"
                   :: "v"(dst), "v"(src) : "memory");
    }
    asm volatile("s_wait_asynccnt 0" ::: "memory");
  }
  __syncthreads();

  int wave = tid >> 5;
  int lane = tid & 31;
  int half = lane >> 4;
  int lm   = lane & 15;
  int m0 = (blockIdx.x * 8 + wave) * 16;

  v8f acc0 = {}, acc1 = {}, acc2 = {}, acc3 = {};
  const __bf16* ap = A + (size_t)(m0 + lm) * K + half * 8;
  const __bf16* bp = sB + lm * K + half * 8;   // LDS
  const int bstep = 16 * K;

  for (int k = 0; k < K; k += 32) {
    union { v16bf v; v8bf h[2]; } af, b0, b1, b2, b3;
    af.h[0] = *(const v8bf*)(ap + k);
    af.h[1] = *(const v8bf*)(ap + k + 16);
    b0.h[0] = *(const v8bf*)(bp + k);
    b0.h[1] = *(const v8bf*)(bp + k + 16);
    b1.h[0] = *(const v8bf*)(bp + bstep + k);
    b1.h[1] = *(const v8bf*)(bp + bstep + k + 16);
    b2.h[0] = *(const v8bf*)(bp + 2 * bstep + k);
    b2.h[1] = *(const v8bf*)(bp + 2 * bstep + k + 16);
    b3.h[0] = *(const v8bf*)(bp + 3 * bstep + k);
    b3.h[1] = *(const v8bf*)(bp + 3 * bstep + k + 16);
    acc0 = __builtin_amdgcn_wmma_f32_16x16x32_bf16(false, af.v, false, b0.v, (short)0, acc0, false, false);
    acc1 = __builtin_amdgcn_wmma_f32_16x16x32_bf16(false, af.v, false, b1.v, (short)0, acc1, false, false);
    acc2 = __builtin_amdgcn_wmma_f32_16x16x32_bf16(false, af.v, false, b2.v, (short)0, acc2, false, false);
    acc3 = __builtin_amdgcn_wmma_f32_16x16x32_bf16(false, af.v, false, b3.v, (short)0, acc3, false, false);
  }

  // C/D layout: VGPR j -> row m0 + j + 8*half, col n0 + nn*16 + lm
  int rb = m0 + half * 8;
#pragma unroll
  for (int j = 0; j < 8; ++j) {
    float* cr = C + (size_t)(rb + j) * N + n0 + lm;
    if (accumulate) {
      cr[0]  += acc0[j]; cr[16] += acc1[j]; cr[32] += acc2[j]; cr[48] += acc3[j];
    } else {
      cr[0]  = acc0[j]; cr[16] = acc1[j]; cr[32] = acc2[j]; cr[48] = acc3[j];
    }
  }
}

// ---------------------------------------------------------------------------
// Depthwise causal conv (k=DC=4, left pad 3) + SiLU.  One thread per (b,t,di).
// Input xc-half of xz (stride 2*DI); outputs f32 and bf16 copies.
// ---------------------------------------------------------------------------
__global__ __launch_bounds__(256) void dwconv_kernel(
    const float* __restrict__ xz,  // (BT, 2*DI), first DI cols are xc
    const float* __restrict__ cw,  // (DI, DC)
    const float* __restrict__ cb,  // (DI)
    float* __restrict__ xc, __bf16* __restrict__ xcb)
{
  int id = blockIdx.x * 256 + threadIdx.x;  // over BT*DI
  int di = id & (DI_ - 1);
  int bt = id / DI_;
  int t  = bt & (T_ - 1);
  const float* w = cw + di * DC_;
  float acc = cb[di];
#pragma unroll
  for (int j = 0; j < DC_; ++j) {
    int tt = t - (DC_ - 1) + j;
    if (tt >= 0) acc += xz[(size_t)(bt - t + tt) * (2 * DI_) + di] * w[j];
  }
  float v = siluf_(acc);
  xc[id]  = v;
  xcb[id] = (__bf16)v;
}

// ---------------------------------------------------------------------------
// dt = softplus(dtl @ W_dt + b_dt). K=8 -> scalar dot. One thread per (bt,di).
// ---------------------------------------------------------------------------
__global__ __launch_bounds__(256) void dtproj_kernel(
    const float* __restrict__ xdbl,  // (BT, XP_) cols 0..7 = dtl
    const float* __restrict__ Wdt,   // (DTR, DI)
    const float* __restrict__ bdt,   // (DI)
    float* __restrict__ dt)          // (BT, DI)
{
  int id = blockIdx.x * 256 + threadIdx.x;
  int di = id & (DI_ - 1);
  int bt = id / DI_;
  const float* dl = xdbl + (size_t)bt * XP_;
  float acc = bdt[di];
#pragma unroll
  for (int r = 0; r < DTR_; ++r) acc += dl[r] * Wdt[r * DI_ + di];
  dt[id] = softplusf_(acc);
}

// ---------------------------------------------------------------------------
// Fused selective-scan with momentum. One wave32 per (b,di); lane == state s
// (DS == 32 == wave width). State h,v live in registers across all T steps;
// y contraction over s is a 5-step shfl_xor wave reduction.
// ---------------------------------------------------------------------------
__global__ __launch_bounds__(256) void scan_kernel(
    const float* __restrict__ dt,    // (BT, DI)
    const float* __restrict__ xc,    // (BT, DI)
    const float* __restrict__ xdbl,  // (BT, XP_): cols 8..39 = B, 40..71 = C
    const float* __restrict__ Alog,  // (DI, DS) for this layer
    const float* __restrict__ alpha, const float* __restrict__ beta,
    float* __restrict__ ys, int layer)
{
  int gw = (blockIdx.x * 256 + threadIdx.x) >> 5;  // global wave 0..B*DI-1
  int s  = threadIdx.x & 31;
  int di = gw & (DI_ - 1);
  int b  = gw >> 8;                                // gw / DI_
  float a  = alpha[layer];
  float bb = beta[layer];
  float A  = -__expf(Alog[di * DS_ + s]);
  float hs = 0.0f, v = 0.0f;
  const float* dtp = dt   + (size_t)b * T_ * DI_ + di;
  const float* xcp = xc   + (size_t)b * T_ * DI_ + di;
  const float* xdp = xdbl + (size_t)b * T_ * XP_;
  float*       yp  = ys   + (size_t)b * T_ * DI_ + di;
  for (int t = 0; t < T_; ++t) {
    float dtv = dtp[(size_t)t * DI_];
    float xcv = xcp[(size_t)t * DI_];
    float Bv  = xdp[(size_t)t * XP_ + DTR_ + s];
    float Cv  = xdp[(size_t)t * XP_ + DTR_ + DS_ + s];
    float dA  = __expf(dtv * A);
    float dBx = dtv * xcv * Bv;
    v  = bb * v + dBx;
    hs = dA * hs + (1.0f - a) * dBx + a * v;
    float y = hs * Cv;
#pragma unroll
    for (int off = 16; off; off >>= 1) y += __shfl_xor(y, off, 32);
    if (s == 0) yp[(size_t)t * DI_] = y;
  }
}

// ---------------------------------------------------------------------------
// Gate: g = (y_scan + D*xc) * silu(z), bf16 out. One thread per (bt,di).
// ---------------------------------------------------------------------------
__global__ __launch_bounds__(256) void gate_kernel(
    const float* __restrict__ ys, const float* __restrict__ xc,
    const float* __restrict__ Dp, const float* __restrict__ xz,
    __bf16* __restrict__ g)
{
  int id = blockIdx.x * 256 + threadIdx.x;
  int di = id & (DI_ - 1);
  int bt = id / DI_;
  float z = xz[(size_t)bt * (2 * DI_) + DI_ + di];
  float y = ys[id] + Dp[di] * xc[id];
  g[id] = (__bf16)(y * siluf_(z));
}

// ---------------------------------------------------------------------------
extern "C" void kernel_launch(void* const* d_in, const int* in_sizes, int n_in,
                              void* d_out, int out_size, void* d_ws, size_t ws_size,
                              hipStream_t stream)
{
  (void)in_sizes; (void)n_in; (void)out_size; (void)ws_size;
  const float* x      = (const float*)d_in[0];
  const float* convfw = (const float*)d_in[1];
  const float* convfb = (const float*)d_in[2];
  const float* bng    = (const float*)d_in[3];
  const float* bnb    = (const float*)d_in[4];
  const float* rmsw   = (const float*)d_in[5];
  const float* Win    = (const float*)d_in[6];
  const float* convw  = (const float*)d_in[7];
  const float* convb  = (const float*)d_in[8];
  const float* Wxp    = (const float*)d_in[9];
  const float* Wdt    = (const float*)d_in[10];
  const float* bdt    = (const float*)d_in[11];
  const float* Alog   = (const float*)d_in[12];
  const float* Dp     = (const float*)d_in[13];
  const float* Wout   = (const float*)d_in[14];
  const float* alpha  = (const float*)d_in[15];
  const float* beta   = (const float*)d_in[16];
  const float* normw  = (const float*)d_in[17];

  char* wsb = (char*)d_ws;
  size_t off = 0;
  auto alloc = [&](size_t bytes) -> void* {
    void* p = wsb + off;
    off = (off + bytes + 255) & ~(size_t)255;
    return p;
  };
  float*  h    = (float*) alloc((size_t)BT_ * D_ * 4);
  __bf16* xn   = (__bf16*)alloc((size_t)BT_ * D_ * 2);
  float*  xz   = (float*) alloc((size_t)BT_ * 2 * DI_ * 4);
  float*  xc   = (float*) alloc((size_t)BT_ * DI_ * 4);
  __bf16* xcb  = (__bf16*)alloc((size_t)BT_ * DI_ * 2);
  float*  xdbl = (float*) alloc((size_t)BT_ * XP_ * 4);
  float*  dt   = (float*) alloc((size_t)BT_ * DI_ * 4);
  float*  ys   = (float*) alloc((size_t)BT_ * DI_ * 4);
  __bf16* g    = (__bf16*)alloc((size_t)BT_ * DI_ * 2);
  __bf16* WinT = (__bf16*)alloc((size_t)(2 * DI_) * D_ * 2);
  __bf16* WxT  = (__bf16*)alloc((size_t)XP_ * DI_ * 2);
  __bf16* WoT  = (__bf16*)alloc((size_t)D_ * DI_ * 2);

  frontend_kernel<<<(BT_ * D_) / 256, 256, 0, stream>>>(x, convfw, convfb, bng, bnb, h);

  for (int l = 0; l < L_; ++l) {
    // weights -> bf16, transposed to (N x K) for contiguous B-fragment loads
    transpose_bf16_kernel<<<(2 * DI_ * D_) / 256, 256, 0, stream>>>(
        Win + (size_t)l * D_ * 2 * DI_, WinT, D_, 2 * DI_);
    transpose_bf16_kernel<<<(XP_ * DI_) / 256, 256, 0, stream>>>(
        Wxp + (size_t)l * DI_ * XPW_, WxT, DI_, XPW_);
    transpose_bf16_kernel<<<(D_ * DI_) / 256, 256, 0, stream>>>(
        Wout + (size_t)l * DI_ * D_, WoT, DI_, D_);

    rmsnorm_bf16_kernel<<<BT_ / 8, 256, 0, stream>>>(h, rmsw + (size_t)l * D_, xn);

    // xz = rmsnorm(x) @ W_in    (8192 x 128) x (128 x 512)
    wmma_gemm_kernel<<<dim3(BT_ / 128, (2 * DI_) / 64), 256, 0, stream>>>(
        xn, WinT, xz, D_, 2 * DI_, 0);

    dwconv_kernel<<<(BT_ * DI_) / 256, 256, 0, stream>>>(
        xz, convw + (size_t)l * DI_ * DC_, convb + (size_t)l * DI_, xc, xcb);

    // xdbl = xc @ W_xproj       (8192 x 256) x (256 x 72->128 padded)
    wmma_gemm_kernel<<<dim3(BT_ / 128, XP_ / 64), 256, 0, stream>>>(
        xcb, WxT, xdbl, DI_, XP_, 0);

    dtproj_kernel<<<(BT_ * DI_) / 256, 256, 0, stream>>>(
        xdbl, Wdt + (size_t)l * DTR_ * DI_, bdt + (size_t)l * DI_, dt);

    scan_kernel<<<(B_ * DI_) / 8, 256, 0, stream>>>(
        dt, xc, xdbl, Alog + (size_t)l * DI_ * DS_, alpha, beta, ys, l);

    gate_kernel<<<(BT_ * DI_) / 256, 256, 0, stream>>>(
        ys, xc, Dp + (size_t)l * DI_, xz, g);

    // h += g @ W_out            (8192 x 256) x (256 x 128), residual accumulate
    wmma_gemm_kernel<<<dim3(BT_ / 128, D_ / 64), 256, 0, stream>>>(
        g, WoT, h, DI_, D_, 1);
  }

  rmsnorm_out_kernel<<<BT_ / 8, 256, 0, stream>>>(h, normw, (float*)d_out);
}
